// FpsLoss_51024211476528
// MI455X (gfx1250) — compile-verified, matrix operand order checked by reading, not run on hardware
//
#include <hip/hip_runtime.h>
#include <math.h>

typedef __attribute__((ext_vector_type(16))) _Float16 v16h;
typedef __attribute__((ext_vector_type(8)))  _Float16 v8h;
typedef __attribute__((ext_vector_type(8)))  float    v8f;

#define DDIM   256
#define BROWS  4096
#define TWOB   8192
#define NTILES 512            // 8192 / 16 column tiles
#define CSPLIT 4              // column chunks per row-strip group
#define CHUNK  (NTILES / CSPLIT)
#define LDS_STRIDE 264        // 256 + 8 halves pad -> conflict-free ds_load_b128
// rows pre-scaled by sqrt(10 * log2(e)): WMMA output = sim/tau * log2(e),
// so raw v_exp_f32 / v_log_f32 (base-2) apply directly; convert to nats at the end.
#define ROW_SCALE 3.7990197f      // sqrt(10 * 1.4426950408889634)
#define LN2       0.69314718055994531f

union FragU { v16h v; v8h h[2]; };

// ---------------- Kernel 0: zero accumulator + per-row sums ----------------
__global__ __launch_bounds__(256) void zero_kernel(float* __restrict__ acc,
                                                   float* __restrict__ rowsum) {
    int i = blockIdx.x * 256 + threadIdx.x;
    if (i < TWOB) rowsum[i] = 0.0f;
    if (i == 0) acc[0] = 0.0f;
}

// ---------------- Kernel 1: L2 normalize rows, write scaled f16 Z ----------------
__global__ __launch_bounds__(256) void norm_kernel(const float* __restrict__ fst,
                                                   const float* __restrict__ snd,
                                                   _Float16* __restrict__ zh) {
    int row = blockIdx.x;
    const float* src = (row < BROWS) ? (fst + (size_t)row * DDIM)
                                     : (snd + (size_t)(row - BROWS) * DDIM);
    float x = src[threadIdx.x];
    __shared__ float red[256];
    red[threadIdx.x] = x * x;
    __syncthreads();
    for (int s = 128; s > 0; s >>= 1) {
        if (threadIdx.x < s) red[threadIdx.x] += red[threadIdx.x + s];
        __syncthreads();
    }
    float scale = ROW_SCALE / fmaxf(sqrtf(red[0]), 1e-6f);
    zh[(size_t)row * DDIM + threadIdx.x] = (_Float16)(x * scale);
}

// ---------------- Kernel 2: partial masked sum(2^s) rows via WMMA ----------------
// Block = 256 threads (8 waves). blockIdx -> (rowgroup, column chunk).
// Wave w owns 16-row strip; all 8 waves share a double-buffered LDS B tile.
__global__ __launch_bounds__(256) void lse_kernel(const _Float16* __restrict__ zh,
                                                  float* __restrict__ rowsum) {
    __shared__ _Float16 bs[2][16 * LDS_STRIDE];

    const int lane  = threadIdx.x & 31;
    const int wave  = threadIdx.x >> 5;
    const int half  = lane >> 4;       // 0: lanes 0-15, 1: lanes 16-31
    const int l15   = lane & 15;
    const int rgrp  = blockIdx.x >> 2;           // 0..63
    const int c0    = (blockIdx.x & (CSPLIT - 1)) * CHUNK;  // first column tile
    const int m0    = (rgrp * 8 + wave) * 16;    // strip base row (M)

    // Cache the full A strip: 8 K-fragments of 16x32 f16 (64 VGPRs).
    v16h afrag[8];
    {
        const _Float16* arow = zh + (size_t)(m0 + l15) * DDIM;
        #pragma unroll
        for (int ks = 0; ks < 8; ++ks) {
            const _Float16* p = arow + ks * 32 + half * 8;
            FragU u;
            u.h[0] = *(const v8h*)(p);        // VGPRs 0-3: K = base .. base+7
            u.h[1] = *(const v8h*)(p + 16);   // VGPRs 4-7: K = base+16 .. base+23
            afrag[ks] = u.v;
        }
    }

    float sumexp[8];
    #pragma unroll
    for (int r = 0; r < 8; ++r) sumexp[r] = 0.0f;

    const int srow = threadIdx.x >> 4;            // staging: row 0..15
    const int sk   = (threadIdx.x & 15) * 16;     // staging: 16-half chunk

    // Prologue: stage first tile of this chunk into buffer 0.
    {
        const v8h* src = (const v8h*)(zh + (size_t)(c0 * 16 + srow) * DDIM + sk);
        v8h* dst = (v8h*)(&bs[0][srow * LDS_STRIDE + sk]);
        dst[0] = src[0];
        dst[1] = src[1];
    }
    __syncthreads();

    for (int it = 0; it < CHUNK; ++it) {
        const int nt  = c0 + it;
        const int cur = it & 1;

        // Issue next tile's global loads early; latency hidden by the WMMA chain.
        const int nn = (it + 1 < CHUNK) ? nt + 1 : nt;   // clamped (uniform flow)
        const v8h* src = (const v8h*)(zh + (size_t)(nn * 16 + srow) * DDIM + sk);
        v8h x0 = src[0];
        v8h x1 = src[1];
        __builtin_prefetch(zh + (size_t)((nn + 1) * 16 + srow) * DDIM + sk, 0, 0);

        // 16x16 tile: D = 256 as 8 chained K=32 WMMAs, B fragments from LDS.
        v8f c = {};
        #pragma unroll
        for (int ks = 0; ks < 8; ++ks) {
            const _Float16* p = &bs[cur][l15 * LDS_STRIDE + ks * 32 + half * 8];
            FragU u;
            u.h[0] = *(const v8h*)(p);
            u.h[1] = *(const v8h*)(p + 16);
            c = __builtin_amdgcn_wmma_f32_16x16x32_f16(
                    false, afrag[ks], false, u.v, (short)0, c, false, false);
        }

        const bool diagTile = (nt * 16 == m0);
        #pragma unroll
        for (int r = 0; r < 8; ++r) {
            // element (M = r + 8*half, N = l15); c is already the base-2 exponent,
            // bounded in [-14.44, 14.44] -> raw v_exp_f32, no subnormal fixup needed.
            float e = __builtin_amdgcn_exp2f(c[r]);
            bool isDiag = diagTile && (l15 == (r + (half << 3)));
            sumexp[r] += isDiag ? 0.0f : e;
        }

        // Store the staged next tile into the other buffer, then one barrier.
        v8h* dst = (v8h*)(&bs[cur ^ 1][srow * LDS_STRIDE + sk]);
        dst[0] = x0;
        dst[1] = x1;
        __syncthreads();
    }

    // Reduce each accumulator across the 16 lanes of its half-wave.
    #pragma unroll
    for (int r = 0; r < 8; ++r) {
        float s = sumexp[r];
        s += __shfl_xor(s, 8, 16);
        s += __shfl_xor(s, 4, 16);
        s += __shfl_xor(s, 2, 16);
        s += __shfl_xor(s, 1, 16);
        sumexp[r] = s;
    }
    // lane 0 holds rows m0+0..7, lane 16 holds rows m0+8..15
    if (l15 == 0) {
        #pragma unroll
        for (int r = 0; r < 8; ++r)
            atomicAdd(&rowsum[m0 + r + (half << 3)], sumexp[r]);
    }
}

// ---------------- Kernel 3: per-row  log2(rowsum) - pos2  ----------------
__global__ __launch_bounds__(256) void posfin_kernel(const _Float16* __restrict__ zh,
                                                     const float* __restrict__ rowsum,
                                                     float* __restrict__ acc) {
    int row  = blockIdx.x * 8 + (threadIdx.x >> 5);
    int lane = threadIdx.x & 31;
    int j    = (row + BROWS) & (TWOB - 1);
    const _Float16* a = zh + (size_t)row * DDIM + lane * 8;
    const _Float16* b = zh + (size_t)j   * DDIM + lane * 8;
    float d = 0.0f;   // product of two ROW_SCALE-scaled rows = sim/tau * log2(e)
    #pragma unroll
    for (int k = 0; k < 8; ++k) d += (float)a[k] * (float)b[k];
    d += __shfl_xor(d, 16, 32);
    d += __shfl_xor(d, 8, 32);
    d += __shfl_xor(d, 4, 32);
    d += __shfl_xor(d, 2, 32);
    d += __shfl_xor(d, 1, 32);
    if (lane == 0)
        atomicAdd(acc, __builtin_amdgcn_logf(rowsum[row]) - d);  // v_log_f32 = log2
}

// ---------------- Kernel 4: finalize (convert base-2 -> nats, mean/2) ----------------
__global__ void finalize_kernel(const float* __restrict__ acc, float* __restrict__ out) {
    if (threadIdx.x == 0) out[0] = acc[0] * (LN2 / (float)(2 * TWOB));
}

extern "C" void kernel_launch(void* const* d_in, const int* in_sizes, int n_in,
                              void* d_out, int out_size, void* d_ws, size_t ws_size,
                              hipStream_t stream) {
    const float* fst = (const float*)d_in[0];
    const float* snd = (const float*)d_in[1];
    float* out = (float*)d_out;

    float*    acc    = (float*)d_ws;                           // 1 f32
    float*    rowsum = (float*)((char*)d_ws + 256);            // 8192 f32 = 32 KB
    _Float16* zh     = (_Float16*)((char*)d_ws + 256 + 32768); // 8192*256 f16 = 4 MB

    zero_kernel<<<TWOB / 256, 256, 0, stream>>>(acc, rowsum);
    norm_kernel<<<TWOB, 256, 0, stream>>>(fst, snd, zh);
    lse_kernel<<<(NTILES / 8) * CSPLIT, 256, 0, stream>>>(zh, rowsum);  // 256 blocks
    posfin_kernel<<<TWOB / 8, 256, 0, stream>>>(zh, rowsum, acc);       // 1024 blocks
    finalize_kernel<<<1, 32, 0, stream>>>(acc, out);
}
